// MultiHeadGraphAttentionLayer_28956669509709
// MI455X (gfx1250) — compile-verified, matrix operand order checked by reading, not run on hardware
//
#include <hip/hip_runtime.h>
#include <hip/hip_bf16.h>

// Reference collapses: sum(softmax(e), axis=-1) == 1, so
//   out = alpha * concat_heads(h @ W) + (1 - alpha) * h
// adj is unused. Single [16384,256]x[256,256] GEMM (f16 in, f32 accum via
// v_wmma_f32_16x16x32_f16) + fused residual blend. ~34 MB traffic -> memory
// bound at 23.3 TB/s (~1.5 us); f16 WMMA compute ~0.4 us, far under the floor.

typedef __attribute__((ext_vector_type(16))) _Float16 v16h;
typedef __attribute__((ext_vector_type(8)))  _Float16 v8h;
typedef __attribute__((ext_vector_type(4)))  _Float16 v4h;
typedef __attribute__((ext_vector_type(8)))  float    v8f;
typedef __attribute__((ext_vector_type(4)))  float    v4f;

#define FDIM 256    // F_in == H*D
#define MTOT 16384  // B*N

// ---- prep: h (f32) -> h16 (f16), 4 elems/thread -------------------------
__global__ __launch_bounds__(256) void cvt_h_f16(const float* __restrict__ src,
                                                 _Float16* __restrict__ dst) {
  int i = (blockIdx.x * 256 + threadIdx.x) * 4;
  v4f v = *(const v4f*)(src + i);
  v4h o;
  o[0] = (_Float16)v[0]; o[1] = (_Float16)v[1];
  o[2] = (_Float16)v[2]; o[3] = (_Float16)v[3];
  *(v4h*)(dst + i) = o;
}

// ---- prep: Wt[n][f] = (f16) W[h][f][d], n = h*64 + d --------------------
__global__ __launch_bounds__(256) void prep_w(const float* __restrict__ W,
                                              _Float16* __restrict__ Wt) {
  int idx = blockIdx.x * 256 + threadIdx.x;  // 0..65535 = n*256 + f
  int n = idx >> 8;
  int f = idx & 255;
  int hh = n >> 6;
  int d  = n & 63;
  Wt[idx] = (_Float16)W[((size_t)(hh * 256 + f) << 6) + d];
}

// Fragment load: ISA 16-bit A/B layout. Lane l: row/col = l&15,
// elements 0..7 at K = kb + (l>>4)*8, elements 8..15 at K = kb + 16 + (l>>4)*8.
// Both chunks are 16B contiguous when K is the fastest-moving axis.
__device__ __forceinline__ v16h load_frag(const _Float16* p) {
  v8h lo = *(const v8h*)p;
  v8h hi = *(const v8h*)(p + 16);
  return __builtin_shufflevector(lo, hi, 0, 1, 2, 3, 4, 5, 6, 7,
                                         8, 9, 10, 11, 12, 13, 14, 15);
}

// ---- GEMM + fused residual blend ----------------------------------------
// Block: 256 threads = 8 waves; tile 32(M) x 256(N).
// Wave (wm 0..1, wn 0..3): one 16-row A strip, four 16x16 N-subtiles.
__global__ __launch_bounds__(256) void gat_gemm(const _Float16* __restrict__ h16,
                                                const _Float16* __restrict__ Wt,
                                                const float*    __restrict__ h,
                                                const float*    __restrict__ alpha_p,
                                                float*          __restrict__ out) {
  const int lane = threadIdx.x & 31;
  const int wave = threadIdx.x >> 5;
  const int wm = wave >> 2;                  // 0..1
  const int wn = wave & 3;                   // 0..3
  const int mbase = blockIdx.x * 32 + wm * 16;
  const int nbase = wn * 64;

  const int lr    = lane & 15;               // row (A) / col (B) within tile
  const int khalf = (lane >> 4) * 8;         // K sub-group offset

  const _Float16* ap = h16 + (size_t)(mbase + lr) * FDIM + khalf;
  const _Float16* bp = Wt  + (size_t)(nbase + lr) * FDIM + khalf;

  v8f acc0 = {}, acc1 = {}, acc2 = {}, acc3 = {};

#pragma unroll
  for (int kb = 0; kb < FDIM; kb += 32) {
    v16h a  = load_frag(ap + kb);
    v16h b0 = load_frag(bp + kb);
    v16h b1 = load_frag(bp + kb + 16 * FDIM);
    v16h b2 = load_frag(bp + kb + 32 * FDIM);
    v16h b3 = load_frag(bp + kb + 48 * FDIM);
    acc0 = __builtin_amdgcn_wmma_f32_16x16x32_f16(false, a, false, b0,
                                                  (short)0, acc0, false, false);
    acc1 = __builtin_amdgcn_wmma_f32_16x16x32_f16(false, a, false, b1,
                                                  (short)0, acc1, false, false);
    acc2 = __builtin_amdgcn_wmma_f32_16x16x32_f16(false, a, false, b2,
                                                  (short)0, acc2, false, false);
    acc3 = __builtin_amdgcn_wmma_f32_16x16x32_f16(false, a, false, b3,
                                                  (short)0, acc3, false, false);
  }

  float alpha = alpha_p[0];
  alpha = fminf(fmaxf(alpha, 0.0f), 1.0f);
  const float beta = 1.0f - alpha;

  // C/D layout: VGPR r -> row = r (lanes 0..15) or r+8 (lanes 16..31); col = l&15.
  const int rbase = mbase + ((lane >> 4) << 3);
  const int cbase = nbase + lr;
#pragma unroll
  for (int r = 0; r < 8; ++r) {
    const size_t ro = (size_t)(rbase + r) * FDIM;
    out[ro + cbase +  0] = alpha * acc0[r] + beta * h[ro + cbase +  0];
    out[ro + cbase + 16] = alpha * acc1[r] + beta * h[ro + cbase + 16];
    out[ro + cbase + 32] = alpha * acc2[r] + beta * h[ro + cbase + 32];
    out[ro + cbase + 48] = alpha * acc3[r] + beta * h[ro + cbase + 48];
  }
}

extern "C" void kernel_launch(void* const* d_in, const int* in_sizes, int n_in,
                              void* d_out, int out_size, void* d_ws, size_t ws_size,
                              hipStream_t stream) {
  const float* h   = (const float*)d_in[0];
  // d_in[1] = adj: unused by the reference (drop_edge identity, then dead).
  const float* W   = (const float*)d_in[2];
  const float* alp = (const float*)d_in[3];
  float* out = (float*)d_out;

  _Float16* h16 = (_Float16*)d_ws;                                   // 8 MB
  _Float16* Wt  = (_Float16*)((char*)d_ws +
                              (size_t)MTOT * FDIM * sizeof(_Float16));  // +128 KB

  cvt_h_f16<<<(MTOT * FDIM) / 1024, 256, 0, stream>>>(h, h16);
  prep_w<<<(FDIM * FDIM) / 256, 256, 0, stream>>>(W, Wt);
  gat_gemm<<<MTOT / 32, 256, 0, stream>>>(h16, Wt, h, alp, out);
}
